// Block_30262339567972
// MI455X (gfx1250) — matrix operand order repaired and checked by reading, not compiled, not run on hardware
//
#include <hip/hip_runtime.h>
#include <hip/hip_bf16.h>
#include <stdint.h>

typedef _Float16 f16;
typedef __attribute__((ext_vector_type(16))) _Float16 v16h;
typedef __attribute__((ext_vector_type(8)))  _Float16 v8h;
typedef __attribute__((ext_vector_type(8)))  float    v8f;
typedef __attribute__((ext_vector_type(4)))  int      v4i;

union V16 { v16h v; v8h h[2]; };

// ---------------------------------------------------------------------------
// CDNA5 async global->LDS copies (ASYNCcnt path); signature probe-confirmed:
// (global int4*, local int4*, imm offset, imm cpol).
// ---------------------------------------------------------------------------
#if defined(__has_builtin)
#if __has_builtin(__builtin_amdgcn_global_load_async_to_lds_b128) && \
    __has_builtin(__builtin_amdgcn_s_wait_asynccnt)
#define HAVE_ASYNC_LDS 1
#endif
#endif
#ifndef HAVE_ASYNC_LDS
#define HAVE_ASYNC_LDS 0
#endif

#if HAVE_ASYNC_LDS
typedef __attribute__((address_space(1))) v4i as1_v4i;
typedef __attribute__((address_space(3))) v4i as3_v4i;
#endif

// copy 16 bytes (8 f16) from global to LDS, per lane
__device__ __forceinline__ void async_copy16(const f16* g, f16* l) {
#if HAVE_ASYNC_LDS
    __builtin_amdgcn_global_load_async_to_lds_b128((as1_v4i*)g, (as3_v4i*)l, 0, 0);
#else
    *(v8h*)l = *(const v8h*)g;
#endif
}

__device__ __forceinline__ void async_join() {
#if HAVE_ASYNC_LDS
    __builtin_amdgcn_s_wait_asynccnt(0);
#endif
}

// ---------------------------------------------------------------------------
// WMMA helpers (CDNA5 16x16x32 f16 -> f32). RA = reuse-A hint: set when the
// NEXT wmma uses the same A fragment register.
// ---------------------------------------------------------------------------
template <bool RA>
__device__ __forceinline__ v8f wmma32(v16h a, v16h b, v8f c) {
    return __builtin_amdgcn_wmma_f32_16x16x32_f16(false, a, false, b, (short)0, c,
                                                  RA, false);
}

// A fragment: 16x32 f16, row-major tile in LDS (stride ld, ld*2 % 16 == 0).
// lane m = lane&15; lanes 0-15 hold K {0..7,16..23}, lanes 16-31 {8..15,24..31}
__device__ __forceinline__ v16h load_frag_a(const f16* base, int ld, int row0, int k0) {
    const int lane = threadIdx.x & 31;
    const int m = lane & 15;
    const int koff = (lane & 16) ? 8 : 0;
    const f16* p = base + (row0 + m) * ld + k0 + koff;
    V16 u;
    u.h[0] = *(const v8h*)(p);
    u.h[1] = *(const v8h*)(p + 16);
    return u.v;
}

// B fragment from a TRANSPOSED tile Bt[n][k]: lane n = lane&15;
// lanes 0-15 hold K {0..7,16..23}, lanes 16-31 {8..15,24..31} (matches A).
__device__ __forceinline__ v16h load_frag_bt(const f16* baseT, int ld, int n0, int k0) {
    const int lane = threadIdx.x & 31;
    const int n = lane & 15;
    const int kr = (lane & 16) ? 8 : 0;
    const f16* p = baseT + (n0 + n) * ld + k0 + kr;
    V16 u;
    u.h[0] = *(const v8h*)(p);
    u.h[1] = *(const v8h*)(p + 16);
    return u.v;
}

// ---------------------------------------------------------------------------
// Tiled transpose + fp32->f16: src[K][N] -> dst[N][K]. Done ONCE per call so
// the GEMM inner loop never transposes weights.
// block (32,8), grid (N/32, K/32).
// ---------------------------------------------------------------------------
__global__ __launch_bounds__(256) void transpose_cvt(const float* __restrict__ src,
                                                     f16* __restrict__ dst,
                                                     int K, int N) {
    __shared__ float tile[32][33];
    const int n0 = blockIdx.x * 32;
    const int k0 = blockIdx.y * 32;
    const int tx = threadIdx.x, ty = threadIdx.y;
#pragma unroll
    for (int i = 0; i < 4; ++i) {
        const int k = ty + i * 8;
        tile[k][tx] = src[(size_t)(k0 + k) * N + n0 + tx];
    }
    __syncthreads();
#pragma unroll
    for (int i = 0; i < 4; ++i) {
        const int n = ty + i * 8;
        dst[(size_t)(n0 + n) * K + k0 + tx] = (f16)tile[tx][n];
    }
}

// ---------------------------------------------------------------------------
// LayerNorm: one 256-thread block per row of C=1024, f32 in, f16 out
// ---------------------------------------------------------------------------
__global__ __launch_bounds__(256) void ln_kernel(const float* __restrict__ x,
                                                 const float* __restrict__ g,
                                                 const float* __restrict__ b,
                                                 f16* __restrict__ out, int C) {
    __shared__ float rs[256];
    __shared__ float rq[256];
    const int row = blockIdx.x;
    const int t = threadIdx.x;
    const float* xr = x + (size_t)row * C;
    float s = 0.f, q = 0.f;
    for (int i = t; i < C; i += 256) {
        float v = xr[i];
        s += v;
        q += v * v;
    }
    rs[t] = s; rq[t] = q;
    __syncthreads();
    for (int off = 128; off > 0; off >>= 1) {
        if (t < off) { rs[t] += rs[t + off]; rq[t] += rq[t + off]; }
        __syncthreads();
    }
    const float mu = rs[0] / (float)C;
    const float var = rq[0] / (float)C - mu * mu;
    const float r = rsqrtf(var + 1e-5f);
    f16* orow = out + (size_t)row * C;
    for (int i = t; i < C; i += 256)
        orow[i] = (f16)((xr[i] - mu) * r * g[i] + b[i]);
}

// ---------------------------------------------------------------------------
// GEMM: out[M,N] = A[M,K](f16, row major) x W^T[N][K](f16, pre-transposed)
// + bias, optional ReLU, optional f32 residual add (f32 out) else f16 out.
// QKV mode: columns ng >= twoC (the V projection) are scattered once into a
// pre-transposed vT[C][M] buffer so attention never transposes V.
// Block tile 64x128, wave tile 32x32, BK=64, stride-72 LDS tiles.
// ---------------------------------------------------------------------------
template <bool RELU, bool RESID, bool QKV>
__global__ __launch_bounds__(256) void gemm_kernel(
    const f16* __restrict__ A, const f16* __restrict__ Wt,
    const float* __restrict__ bias, const float* __restrict__ resid,
    float* __restrict__ outF, f16* __restrict__ outH, f16* __restrict__ vT,
    int twoC, int M, int N, int K) {
    __shared__ f16 sA[64 * 72];
    __shared__ f16 sBt[128 * 72];

    const int t = threadIdx.x;
    const int wave = t >> 5;
    const int wm = wave & 1;        // 2 wave rows
    const int wn = wave >> 1;       // 4 wave cols
    const int m0 = blockIdx.y * 64;
    const int n0 = blockIdx.x * 128;

    v8f acc00 = {}, acc01 = {}, acc10 = {}, acc11 = {};

    const int arow = t >> 2;
    const int ach = (t & 3) * 16;
    const int brow = t >> 1;
    const int bch = (t & 1) * 32;

    const int nkt = K >> 6;
    for (int kt = 0; kt < nkt; ++kt) {
        // ---- stage A tile 64x64 (row major), 32B per thread ----
        {
            const f16* gp = A + (size_t)(m0 + arow) * K + kt * 64 + ach;
            f16* lp = sA + arow * 72 + ach;
            async_copy16(gp, lp);
            async_copy16(gp + 8, lp + 8);
        }
        // ---- stage Bt tile 128x64 (already [n][k]), 64B per thread ----
        {
            const f16* gp = Wt + (size_t)(n0 + brow) * K + kt * 64 + bch;
            f16* lp = sBt + brow * 72 + bch;
            async_copy16(gp, lp);
            async_copy16(gp + 8, lp + 8);
            async_copy16(gp + 16, lp + 16);
            async_copy16(gp + 24, lp + 24);
        }
        if (kt + 1 < nkt) {  // global_prefetch_b8 for the next K tile
            __builtin_prefetch((const void*)(A + (size_t)(m0 + arow) * K +
                                             (kt + 1) * 64 + ach), 0, 1);
            __builtin_prefetch((const void*)(Wt + (size_t)(n0 + brow) * K +
                                             (kt + 1) * 64 + bch), 0, 1);
        }
        async_join();
        __syncthreads();

#pragma unroll
        for (int ks = 0; ks < 2; ++ks) {
            const int kb = ks * 32;
            v16h a0 = load_frag_a(sA, 72, wm * 32, kb);
            v16h a1 = load_frag_a(sA, 72, wm * 32 + 16, kb);
            v16h b0 = load_frag_bt(sBt, 72, wn * 32, kb);
            v16h b1 = load_frag_bt(sBt, 72, wn * 32 + 16, kb);
            acc00 = wmma32<true>(a0, b0, acc00);   // next wmma reuses a0
            acc01 = wmma32<false>(a0, b1, acc01);
            acc10 = wmma32<true>(a1, b0, acc10);   // next wmma reuses a1
            acc11 = wmma32<false>(a1, b1, acc11);
        }
        __syncthreads();
    }

    // ---- epilogue: bias (+ relu) (+ residual / V-transpose scatter) ----
    const int lane = t & 31;
    const int nloc = lane & 15;
    const int rshift = (lane & 16) ? 8 : 0;
    v8f accs[4] = {acc00, acc01, acc10, acc11};
#pragma unroll
    for (int su = 0; su < 2; ++su) {
#pragma unroll
        for (int sv = 0; sv < 2; ++sv) {
            v8f a = accs[su * 2 + sv];
#pragma unroll
            for (int r = 0; r < 8; ++r) {
                const int mg = m0 + wm * 32 + su * 16 + r + rshift;
                const int ng = n0 + wn * 32 + sv * 16 + nloc;
                float v = a[r] + bias[ng];
                if (RELU) v = v > 0.f ? v : 0.f;
                if (RESID) {
                    outF[(size_t)mg * N + ng] = resid[(size_t)mg * N + ng] + v;
                } else if (QKV && ng >= twoC) {
                    vT[(size_t)(ng - twoC) * M + mg] = (f16)v;
                } else {
                    outH[(size_t)mg * N + ng] = (f16)v;
                }
            }
        }
    }
}

// ---------------------------------------------------------------------------
// Flash attention: one block per (b, h, 64-row q tile). Causal, D=64.
// Q,K tiles staged row-major ([tok][d] == transposed-B layout for QK^T);
// V staged from the pre-transposed vT[C][M] buffer ([d][tok] rows), so ALL
// tile staging is pure async b128 copies.
// 8 waves: wave = (qi in 0..3) x (dh in 0..1).
// ---------------------------------------------------------------------------
__global__ __launch_bounds__(256) void attn_kernel(const f16* __restrict__ qkv,
                                                   const f16* __restrict__ vT,
                                                   f16* __restrict__ y, int B,
                                                   int T, int C, int H) {
    const int D = 64;
    const int TQ = T / 64;
    const int bh = blockIdx.x / TQ;
    const int qt = blockIdx.x % TQ;
    const int b = bh / H;
    const int h = bh % H;
    const int C3 = 3 * C;
    const size_t Mtok = (size_t)B * T;

    __shared__ f16 sQ[64 * 72];     // [q][d]
    __shared__ f16 sK[64 * 72];     // [kv][d]  == transposed-B layout for QK^T
    __shared__ f16 sVt[64 * 72];    // [d][kv]  == transposed-B layout for P@V
    __shared__ float sS[64 * 65];   // scores f32 (65 stride: conflict-free)
    __shared__ f16 sP[64 * 72];     // probs f16 (A layout source)
    __shared__ float sM[64], sL[64], sAl[64];

    const int t = threadIdx.x;
    const int wave = t >> 5;
    const int qi = wave & 3;
    const int dh = wave >> 2;
    const int lane = t & 31;
    const int row = t >> 2;
    const int ch = (t & 3) * 16;

    // ---- stage Q tile ----
    {
        const f16* gp =
            qkv + (size_t)((size_t)b * T + qt * 64 + row) * C3 + h * D + ch;
        f16* lp = sQ + row * 72 + ch;
        async_copy16(gp, lp);
        async_copy16(gp + 8, lp + 8);
    }
    if (t < 64) { sM[t] = -3.0e38f; sL[t] = 0.f; }
    async_join();
    __syncthreads();

    const v16h aq0 = load_frag_a(sQ, 72, qi * 16, 0);
    const v16h aq1 = load_frag_a(sQ, 72, qi * 16, 32);

    v8f o0 = {}, o1 = {};

    for (int kvt = 0; kvt <= qt; ++kvt) {
        // ---- stage K tile ([kv][d]) and V tile ([d][kv], from vT) ----
        {
            const size_t tok = (size_t)b * T + kvt * 64 + row;
            const f16* gk = qkv + tok * C3 + C + h * D + ch;
            f16* lk = sK + row * 72 + ch;
            async_copy16(gk, lk);
            async_copy16(gk + 8, lk + 8);
            // row = d index, ch = kv chunk
            const f16* gv = vT + (size_t)(h * D + row) * Mtok +
                            (size_t)b * T + kvt * 64 + ch;
            f16* lv = sVt + row * 72 + ch;
            async_copy16(gv, lv);
            async_copy16(gv + 8, lv + 8);
        }
        async_join();
        __syncthreads();

        // ---- S = (Q K^T) * scale, causal mask, stash to LDS ----
#pragma unroll
        for (int nt = 0; nt < 2; ++nt) {
            const int nl = dh * 32 + nt * 16;
            v8f s = {};
            s = wmma32<false>(aq0, load_frag_bt(sK, 72, nl, 0), s);
            s = wmma32<false>(aq1, load_frag_bt(sK, 72, nl, 32), s);
            const int nloc = lane & 15;
            const int rshift = (lane & 16) ? 8 : 0;
#pragma unroll
            for (int r = 0; r < 8; ++r) {
                const int ml = qi * 16 + r + rshift;
                const int qg = qt * 64 + ml;
                const int kg = kvt * 64 + nl + nloc;
                float v = s[r] * 0.125f;  // D^-0.5, D=64
                if (kg > qg) v = -3.0e38f;
                sS[ml * 65 + nl + nloc] = v;
            }
        }
        __syncthreads();

        // ---- online softmax, one thread per q row ----
        if (t < 64) {
            const float* srow = sS + t * 65;
            float mnew = sM[t];
            for (int i = 0; i < 64; ++i) mnew = fmaxf(mnew, srow[i]);
            const float alpha = __expf(sM[t] - mnew);
            float sum = 0.f;
            f16* prow = sP + t * 72;
            for (int i = 0; i < 64; ++i) {
                const float p = __expf(srow[i] - mnew);
                prow[i] = (f16)p;
                sum += p;
            }
            sM[t] = mnew;
            sL[t] = alpha * sL[t] + sum;
            sAl[t] = alpha;
        }
        __syncthreads();

        // ---- rescale O, accumulate P @ V ----
        {
            const int rshift = (lane & 16) ? 8 : 0;
#pragma unroll
            for (int r = 0; r < 8; ++r) {
                const float a = sAl[qi * 16 + r + rshift];
                o0[r] *= a;
                o1[r] *= a;
            }
        }
        const v16h ap0 = load_frag_a(sP, 72, qi * 16, 0);
        const v16h ap1 = load_frag_a(sP, 72, qi * 16, 32);
        // pair by shared A fragment; reuse_a on the first of each pair
        o0 = wmma32<true>(ap0, load_frag_bt(sVt, 72, dh * 32, 0), o0);
        o1 = wmma32<false>(ap0, load_frag_bt(sVt, 72, dh * 32 + 16, 0), o1);
        o0 = wmma32<true>(ap1, load_frag_bt(sVt, 72, dh * 32, 32), o0);
        o1 = wmma32<false>(ap1, load_frag_bt(sVt, 72, dh * 32 + 16, 32), o1);
        __syncthreads();
    }

    // ---- normalize and write y[b, q, h*D + d] as f16 ----
    {
        const int nloc = lane & 15;
        const int rshift = (lane & 16) ? 8 : 0;
#pragma unroll
        for (int r = 0; r < 8; ++r) {
            const int ml = qi * 16 + r + rshift;
            const float inv = 1.f / sL[ml];
            const size_t rowg = (size_t)b * T + qt * 64 + ml;
            y[rowg * C + h * D + dh * 32 + nloc] = (f16)(o0[r] * inv);
            y[rowg * C + h * D + dh * 32 + 16 + nloc] = (f16)(o1[r] * inv);
        }
    }
}

// ---------------------------------------------------------------------------
// Host launcher
// ---------------------------------------------------------------------------
extern "C" void kernel_launch(void* const* d_in, const int* in_sizes, int n_in,
                              void* d_out, int out_size, void* d_ws,
                              size_t ws_size, hipStream_t stream) {
    (void)in_sizes; (void)n_in; (void)out_size; (void)ws_size;
    const int B = 4, T = 2048, C = 1024, H = 16;
    const int M = B * T;  // 8192 rows

    const float* x      = (const float*)d_in[0];
    const float* ln1_g  = (const float*)d_in[1];
    const float* ln1_b  = (const float*)d_in[2];
    const float* attn_w = (const float*)d_in[3];
    const float* attn_b = (const float*)d_in[4];
    const float* proj_w = (const float*)d_in[5];
    const float* proj_b = (const float*)d_in[6];
    const float* ln2_g  = (const float*)d_in[7];
    const float* ln2_b  = (const float*)d_in[8];
    const float* fc1_w  = (const float*)d_in[9];
    const float* fc1_b  = (const float*)d_in[10];
    const float* fc2_w  = (const float*)d_in[11];
    const float* fc2_b  = (const float*)d_in[12];
    float* out = (float*)d_out;

    char* ws = (char*)d_ws;
    size_t off = 0;
    auto wsalloc = [&](size_t bytes) -> char* {
        char* p = ws + off;
        off += (bytes + 255) & ~(size_t)255;
        return p;
    };
    // pre-transposed f16 weights: Wt[N][K]
    f16* wAttnT = (f16*)wsalloc((size_t)C * 3 * C * sizeof(f16));
    f16* wProjT = (f16*)wsalloc((size_t)C * C * sizeof(f16));
    f16* wFc1T  = (f16*)wsalloc((size_t)C * 4 * C * sizeof(f16));
    f16* wFc2T  = (f16*)wsalloc((size_t)4 * C * C * sizeof(f16));
    f16* hbuf   = (f16*)wsalloc((size_t)M * C * sizeof(f16));
    f16* qkvbuf = (f16*)wsalloc((size_t)M * 3 * C * sizeof(f16));
    f16* vTbuf  = (f16*)wsalloc((size_t)C * M * sizeof(f16));  // V, [h*D+d][tok]
    f16* ybuf   = (f16*)wsalloc((size_t)M * C * sizeof(f16));
    float* x1   = (float*)wsalloc((size_t)M * C * sizeof(float));
    f16* act    = (f16*)wsalloc((size_t)M * 4 * C * sizeof(f16));

    auto tcvt = [&](const float* s, f16* d, int K, int N) {
        transpose_cvt<<<dim3(N / 32, K / 32), dim3(32, 8), 0, stream>>>(s, d, K, N);
    };
    tcvt(attn_w, wAttnT, C, 3 * C);
    tcvt(proj_w, wProjT, C, C);
    tcvt(fc1_w, wFc1T, C, 4 * C);
    tcvt(fc2_w, wFc2T, 4 * C, C);

    // h = ln1(x)
    ln_kernel<<<M, 256, 0, stream>>>(x, ln1_g, ln1_b, hbuf, C);
    // qkv = h @ attn_w + attn_b  (q,k -> qkvbuf f16; v -> vTbuf transposed)
    gemm_kernel<false, false, true><<<dim3(3 * C / 128, M / 64), 256, 0, stream>>>(
        hbuf, wAttnT, attn_b, nullptr, nullptr, qkvbuf, vTbuf, 2 * C, M, 3 * C, C);
    // y = causal flash attention (f16 out)
    attn_kernel<<<B * H * (T / 64), 256, 0, stream>>>(qkvbuf, vTbuf, ybuf, B, T, C, H);
    // x1 = x + y @ proj_w + proj_b  (f32 out)
    gemm_kernel<false, true, false><<<dim3(C / 128, M / 64), 256, 0, stream>>>(
        ybuf, wProjT, proj_b, x, x1, nullptr, nullptr, 0, M, C, C);
    // h = ln2(x1)
    ln_kernel<<<M, 256, 0, stream>>>(x1, ln2_g, ln2_b, hbuf, C);
    // act = relu(h @ fc1_w + fc1_b)  (f16 out)
    gemm_kernel<true, false, false><<<dim3(4 * C / 128, M / 64), 256, 0, stream>>>(
        hbuf, wFc1T, fc1_b, nullptr, nullptr, act, nullptr, 0, M, 4 * C, C);
    // out = x1 + act @ fc2_w + fc2_b  (f32 out)
    gemm_kernel<false, true, false><<<dim3(C / 128, M / 64), 256, 0, stream>>>(
        act, wFc2T, fc2_b, x1, out, nullptr, nullptr, 0, M, C, 4 * C);
}
